// MultiHeadAttention_8160437862703
// MI455X (gfx1250) — matrix operand order repaired and checked by reading, not compiled, not run on hardware
//
#include <hip/hip_runtime.h>

typedef __attribute__((ext_vector_type(16))) _Float16 v16h;
typedef __attribute__((ext_vector_type(8)))  _Float16 v8h;
typedef __attribute__((ext_vector_type(8)))  float    v8f;

#define BB 4
#define SS 2048
#define DD 1024
#define HH 16
#define DKK 64
#define MM (BB*SS)

// ---------------------------------------------------------------- f32 -> f16
__global__ void cvt_f32_f16(const float* __restrict__ src,
                            _Float16* __restrict__ dst, int n) {
  int i = blockIdx.x * blockDim.x + threadIdx.x;
  if (i < n) dst[i] = (_Float16)src[i];
}

// Per-lane async copy of 32 contiguous bytes (2 x B128) global -> LDS.
// The LDS address is the low 32 bits of the flat address, so truncating the generic
// shared-memory pointer yields the ds-usable LDS byte address.
__device__ __forceinline__ void async_copy32(const _Float16* gsrc,
                                             _Float16* lds_dst) {
  unsigned d = (unsigned)(size_t)lds_dst;
  asm volatile("global_load_async_to_lds_b128 %0, %1, off"
               :: "v"(d), "v"(gsrc) : "memory");
  asm volatile("global_load_async_to_lds_b128 %0, %1, off offset:16"
               :: "v"(d), "v"(gsrc) : "memory");
}

// ---------------------------------------------------------------- GEMM
// C[m,n] = sum_k A[m,k] * Bw[n,k]   (Bw is [N][K] row-major == torch Linear W)
// mode 0: store f16 at [b,h,s,dk]      (Q, K)
// mode 1: store f16 at [b,h,dk,s]      (V transposed for PV B-fragments)
// mode 2: store f32 at [m, DD]         (final output projection)
__global__ __launch_bounds__(256)
void gemm_wmma(const _Float16* __restrict__ A, const _Float16* __restrict__ Bw,
               void* __restrict__ out, int mode) {
  constexpr int KD = DD;   // K = 1024 at every call site
  // double-buffered staging tiles, filled by ASYNCcnt-tracked async loads
  __shared__ __align__(32) _Float16 lsA[2][128 * 32];
  __shared__ __align__(32) _Float16 lsB[2][128 * 32];

  const int tid  = threadIdx.x;
  const int lane = tid & 31;
  const int l15  = lane & 15;
  const int hg   = lane >> 4;          // half-group (wave32)
  const int wave = tid >> 5;           // 0..7
  const int wm   = (wave & 1) * 64;    // wave tile M offset
  const int wn   = (wave >> 1) * 32;   // wave tile N offset
  const int n0   = blockIdx.x * 128;
  const int m0   = blockIdx.y * 128;

  v8f acc[4][2] = {};

  const int ldr = tid >> 1;            // 0..127 (row within tile)
  const int lc  = (tid & 1) * 16;      // 0 or 16 (col within 32-wide K slab)
  const int st  = ldr * 32 + lc;       // staging offset within a tile
  const _Float16* gA = A  + (size_t)(m0 + ldr) * KD + lc;
  const _Float16* gB = Bw + (size_t)(n0 + ldr) * KD + lc;

  // 8 WMMAs on the tile sitting in buffer `bufsel`
  auto compute_tile = [&](int bufsel) {
    // B fragments: lane -> N = l15 (+16 per sub), K = hg*16 + i (contiguous)
    v16h bf[2];
#pragma unroll
    for (int ns = 0; ns < 2; ++ns)
      bf[ns] = *(const v16h*)&lsB[bufsel][(wn + ns * 16 + l15) * 32 + hg * 16];

#pragma unroll
    for (int ms = 0; ms < 4; ++ms) {
      // A fragment: lane -> M = l15, K = hg*16 + i (contiguous)
      const _Float16* pa = &lsA[bufsel][(wm + ms * 16 + l15) * 32 + hg * 16];
      v8h a0 = *(const v8h*)pa;
      v8h a1 = *(const v8h*)(pa + 8);
      v16h af;
#pragma unroll
      for (int i = 0; i < 8; ++i) { af[i] = a0[i]; af[i + 8] = a1[i]; }
#pragma unroll
      for (int ns = 0; ns < 2; ++ns)
        acc[ms][ns] = __builtin_amdgcn_wmma_f32_16x16x32_f16(
            false, af, false, bf[ns], (short)0, acc[ms][ns], false, false);
    }
  };

  // prologue: async-stage tile 0 into buffer 0 (4 asyncs per wave)
  async_copy32(gA, &lsA[0][st]);
  async_copy32(gB, &lsB[0][st]);

  // steady state: a next tile always exists -> branch-free pipeline body.
  // unroll(1): keep ONE body so accumulators keep stable register homes
  // (2x unrolling made the allocator shuffle acc between register sets).
  int buf = 0;
#pragma unroll 1
  for (int k0 = 0; k0 < KD - 32; k0 += 32) {
    // issue next tile into the other buffer, then wait only for the CURRENT
    // tile (async loads complete in order; the 4 newer stay in flight)
    async_copy32(gA + k0 + 32, &lsA[buf ^ 1][st]);
    async_copy32(gB + k0 + 32, &lsB[buf ^ 1][st]);
    asm volatile("s_wait_asynccnt 0x4" ::: "memory");
    __syncthreads();
    compute_tile(buf);
    __syncthreads();   // everyone done reading buf before it is overwritten
    buf ^= 1;
  }
  // peeled last tile: nothing left to issue, drain ASYNCcnt fully
  asm volatile("s_wait_asynccnt 0x0" ::: "memory");
  __syncthreads();
  compute_tile(buf);

  // epilogue
#pragma unroll
  for (int ms = 0; ms < 4; ++ms)
#pragma unroll
    for (int ns = 0; ns < 2; ++ns)
#pragma unroll
      for (int r = 0; r < 8; ++r) {
        int m = m0 + wm + ms * 16 + r + 8 * hg;   // C layout: row = r + 8*hg
        int n = n0 + wn + ns * 16 + l15;          //           col = l15
        float v = acc[ms][ns][r];
        if (mode == 2) {
          ((float*)out)[(size_t)m * DD + n] = v;
        } else {
          int b = m >> 11, s = m & (SS - 1);
          int h = n >> 6,  c = n & 63;
          size_t bh = (size_t)(b * HH + h);
          _Float16 hv = (_Float16)v;
          if (mode == 0)
            ((_Float16*)out)[(bh * SS + s) * DKK + c] = hv;   // [b,h,s,dk]
          else
            ((_Float16*)out)[(bh * DKK + c) * SS + s] = hv;   // [b,h,dk,s]
        }
      }
}

// ---------------------------------------------------------------- flash attn
// One wave per (b,h, 16-query tile). Online softmax over causal key range.
__global__ __launch_bounds__(32)
void flash_attn(const _Float16* __restrict__ Qh, const _Float16* __restrict__ Kh,
                const _Float16* __restrict__ Vt, const float* __restrict__ pad,
                _Float16* __restrict__ Ctx) {
  __shared__ __align__(32) _Float16 lp[16 * 32];

  const int lane = threadIdx.x;
  const int l15  = lane & 15;
  const int hg   = lane >> 4;
  const int qt   = blockIdx.x & (SS / 16 - 1);
  const int bh   = blockIdx.x >> 7;          // SS/16 == 128
  const int b    = bh >> 4;                  // HH == 16
  const int h    = bh & 15;
  const int qbase = qt * 16;

  const _Float16* Qp = Qh + (size_t)bh * SS * DKK;   // [s][dk]
  const _Float16* Kp = Kh + (size_t)bh * SS * DKK;   // [s][dk]
  const _Float16* Vp = Vt + (size_t)bh * DKK * SS;   // [dk][s]
  const float* padp  = pad + (size_t)b * SS;

  // Q A-fragments for dk chunks {0..31},{32..63}
  v16h qf[2];
#pragma unroll
  for (int dc = 0; dc < 2; ++dc) {
    const _Float16* p = Qp + (size_t)(qbase + l15) * DKK + dc * 32 + hg * 16;
    v8h a0 = *(const v8h*)p;
    v8h a1 = *(const v8h*)(p + 8);
#pragma unroll
    for (int i = 0; i < 8; ++i) { qf[dc][i] = a0[i]; qf[dc][i + 8] = a1[i]; }
  }

  v8f ctx[4] = {};
  float mrow[8], lrow[8];
#pragma unroll
  for (int r = 0; r < 8; ++r) { mrow[r] = -3.0e38f; lrow[r] = 0.f; }

  const float scale = 0.125f;               // 1/sqrt(64)
  const int qmax = qbase + 15;

#pragma unroll 1
  for (int kt = 0; kt <= qmax; kt += 32) {
    // prefetch next key tile (global_prefetch_b8; speculative, cheap)
    if (kt + 32 <= qmax) {
      __builtin_prefetch(Kp + (size_t)(kt + 32 + l15) * DKK, 0, 1);
      __builtin_prefetch(Vp + (size_t)l15 * SS + kt + 32, 0, 1);
    }

    // ---- scores: 16 q x 32 k, dk=64 accumulated over two K=32 WMMAs
    v8f s0 = {}, s1 = {};
#pragma unroll
    for (int dc = 0; dc < 2; ++dc) {
      v16h kf0 = *(const v16h*)(Kp + (size_t)(kt + l15) * DKK + dc * 32 + hg * 16);
      v16h kf1 = *(const v16h*)(Kp + (size_t)(kt + 16 + l15) * DKK + dc * 32 + hg * 16);
      s0 = __builtin_amdgcn_wmma_f32_16x16x32_f16(false, qf[dc], false, kf0,
                                                  (short)0, s0, false, false);
      s1 = __builtin_amdgcn_wmma_f32_16x16x32_f16(false, qf[dc], false, kf1,
                                                  (short)0, s1, false, false);
    }

    const int c0 = kt + l15, c1 = c0 + 16;
    const float pad0 = padp[c0];
    const float pad1 = padp[c1];

    float alpha[8];
#pragma unroll
    for (int r = 0; r < 8; ++r) {
      int q = qbase + r + 8 * hg;
      float v0 = s0[r] * scale, v1 = s1[r] * scale;
      if (c0 > q || pad0 == 0.f) v0 = -3.0e38f;
      if (c1 > q || pad1 == 0.f) v1 = -3.0e38f;
      float mx = fmaxf(v0, v1);
#pragma unroll
      for (int o = 8; o >= 1; o >>= 1) mx = fmaxf(mx, __shfl_xor(mx, o, 32));
      float mnew = fmaxf(mrow[r], mx);
      float e0 = (v0 <= -1.0e30f) ? 0.f : __expf(v0 - mnew);
      float e1 = (v1 <= -1.0e30f) ? 0.f : __expf(v1 - mnew);
      float rs = e0 + e1;
#pragma unroll
      for (int o = 8; o >= 1; o >>= 1) rs += __shfl_xor(rs, o, 32);
      alpha[r] = __expf(mrow[r] - mnew);
      lrow[r]  = lrow[r] * alpha[r] + rs;
      mrow[r]  = mnew;
      lp[(r + 8 * hg) * 32 + l15]      = (_Float16)e0;   // P tile to LDS
      lp[(r + 8 * hg) * 32 + l15 + 16] = (_Float16)e1;
    }
    __syncthreads();

    // P A-fragment (16 q x 32 k) from LDS
    const _Float16* pp = &lp[l15 * 32 + hg * 16];
    v8h a0 = *(const v8h*)pp;
    v8h a1 = *(const v8h*)(pp + 8);
    v16h pf;
#pragma unroll
    for (int i = 0; i < 8; ++i) { pf[i] = a0[i]; pf[i + 8] = a1[i]; }

    // ctx = ctx*alpha + P @ V  (V columns contiguous via [dk][s] layout)
#pragma unroll
    for (int ns = 0; ns < 4; ++ns) {
#pragma unroll
      for (int r = 0; r < 8; ++r) ctx[ns][r] *= alpha[r];
      v16h vf = *(const v16h*)(Vp + (size_t)(ns * 16 + l15) * SS + kt + hg * 16);
      ctx[ns] = __builtin_amdgcn_wmma_f32_16x16x32_f16(
          false, pf, false, vf, (short)0, ctx[ns], false, false);
    }
    __syncthreads();
  }

  // normalize + store ctx as f16 [b, s, h*64 + c]
#pragma unroll
  for (int ns = 0; ns < 4; ++ns)
#pragma unroll
    for (int r = 0; r < 8; ++r) {
      float inv = lrow[r] > 0.f ? 1.f / lrow[r] : 0.f;
      int s = qbase + r + 8 * hg;
      Ctx[((size_t)b * SS + s) * DD + h * DKK + ns * 16 + l15] =
          (_Float16)(ctx[ns][r] * inv);
    }
}

// ---------------------------------------------------------------- launch
extern "C" void kernel_launch(void* const* d_in, const int* in_sizes, int n_in,
                              void* d_out, int out_size, void* d_ws, size_t ws_size,
                              hipStream_t stream) {
  const float* X   = (const float*)d_in[0];
  const float* pad = (const float*)d_in[1];
  // d_in[2] = num_heads (fixed 16, compiled in)
  const float* WQ  = (const float*)d_in[3];
  const float* WK  = (const float*)d_in[4];
  const float* WV  = (const float*)d_in[5];
  const float* WO  = (const float*)d_in[6];

  char* ws = (char*)d_ws;
  const size_t MB = 1u << 20;
  _Float16* Xh   = (_Float16*)(ws);                 // 16 MB  [M, D]
  _Float16* WQh  = (_Float16*)(ws + 16 * MB);       //  2 MB  [N, K]
  _Float16* WKh  = (_Float16*)(ws + 18 * MB);
  _Float16* WVh  = (_Float16*)(ws + 20 * MB);
  _Float16* WOh  = (_Float16*)(ws + 22 * MB);
  _Float16* Qh   = (_Float16*)(ws + 24 * MB);       // 16 MB  [b,h,s,dk]
  _Float16* Kh   = (_Float16*)(ws + 40 * MB);       // 16 MB  [b,h,s,dk]
  _Float16* Vt   = (_Float16*)(ws + 56 * MB);       // 16 MB  [b,h,dk,s]
  _Float16* Ctxh = (_Float16*)(ws + 72 * MB);       // 16 MB  [M, D]

  const int nX = MM * DD, nW = DD * DD;
  cvt_f32_f16<<<(nX + 255) / 256, 256, 0, stream>>>(X,  Xh,  nX);
  cvt_f32_f16<<<(nW + 255) / 256, 256, 0, stream>>>(WQ, WQh, nW);
  cvt_f32_f16<<<(nW + 255) / 256, 256, 0, stream>>>(WK, WKh, nW);
  cvt_f32_f16<<<(nW + 255) / 256, 256, 0, stream>>>(WV, WVh, nW);
  cvt_f32_f16<<<(nW + 255) / 256, 256, 0, stream>>>(WO, WOh, nW);

  dim3 ggrid(DD / 128, MM / 128);   // (8, 64)
  gemm_wmma<<<ggrid, 256, 0, stream>>>(Xh, WQh, Qh, 0);
  gemm_wmma<<<ggrid, 256, 0, stream>>>(Xh, WKh, Kh, 0);
  gemm_wmma<<<ggrid, 256, 0, stream>>>(Xh, WVh, Vt, 1);

  flash_attn<<<BB * HH * (SS / 16), 32, 0, stream>>>(Qh, Kh, Vt, pad, Ctxh);

  gemm_wmma<<<ggrid, 256, 0, stream>>>(Ctxh, WOh, d_out, 2);
}
